// CriticRNN_14937896255745
// MI455X (gfx1250) — matrix operand order repaired and checked by reading, not compiled, not run on hardware
//
#include <hip/hip_runtime.h>
#include <hip/hip_bf16.h>

// ---------------------------------------------------------------------------
// CriticRNN (GRU) for MI455X / gfx1250: wave32, WMMA bf16, double-buffered
// TDM (tensor_load_to_lds) tile staging overlapped with matrix math.
// T=128, B=128, OBS=512, H=1024.
// ---------------------------------------------------------------------------

#define T_STEPS 128
#define BATCH   128
#define OBS_DIM 512
#define HID     1024

typedef __attribute__((ext_vector_type(16))) __bf16 v16bf;
typedef __attribute__((ext_vector_type(8)))  float  v8f;
typedef unsigned int  v4u __attribute__((ext_vector_type(4)));
typedef int           v8i __attribute__((ext_vector_type(8)));
typedef int           v4i __attribute__((ext_vector_type(4)));
typedef unsigned short ushort_t;

union FragBF { uint4 q[2]; v16bf v; };

__device__ __forceinline__ ushort_t f2bf(float f) {
    unsigned u = __builtin_bit_cast(unsigned, f);
    unsigned r = u + 0x7FFFu + ((u >> 16) & 1u);   // round-to-nearest-even
    return (ushort_t)(r >> 16);
}
__device__ __forceinline__ float bf2f(ushort_t h) {
    return __builtin_bit_cast(float, ((unsigned)h) << 16);
}
__device__ __forceinline__ float sigmoidf_(float x) {
    return 1.0f / (1.0f + __expf(-x));
}

#define BK   32
#define APAD 40   // 32 data + 8 pad ushorts -> 80B pitch (= 16 DW data + 4 DW pad)

// ---------------------------------------------------------------------------
// Tensor Data Mover: 2D bf16 tile into LDS, APAD pitch via D# padding
// (4 DWORDs of pad every 16 DWORDs of data).
// ---------------------------------------------------------------------------
#if defined(__has_builtin)
#  if __has_builtin(__builtin_amdgcn_tensor_load_to_lds)
#    define HAVE_TDM 1
#  endif
#endif

#if defined(HAVE_TDM)
__device__ __forceinline__ void tdm_load_tile_bf16(const ushort_t* gsrc,
                                                   unsigned lds_byte_addr,
                                                   unsigned tensor_d0,   // row length (elems)
                                                   unsigned tensor_d1,   // #rows in tensor
                                                   unsigned tile_d0,     // tile cols (elems)
                                                   unsigned tile_d1,     // tile rows
                                                   unsigned stride_elems)
{
    unsigned long long ga = (unsigned long long)(uintptr_t)gsrc;
    v4u g0;
    g0.x = 0x1u;                                               // count=1, user mode
    g0.y = lds_byte_addr;                                      // lds_addr [63:32]
    g0.z = (unsigned)(ga & 0xffffffffu);                       // global_addr [95:64]
    g0.w = (unsigned)((ga >> 32) & 0x01ffffffu) | (2u << 30);  // addr[56:32] | type=2
    v8i g1;
    g1[0] = (int)((1u << 16)        // data_size = 2 bytes
                | (1u << 20)        // pad_enable
                | (3u << 22)        // pad_interval: 16 DWORDs
                | (3u << 25));      // pad_amount:   4 DWORDs
    g1[1] = (int)((tensor_d0 & 0xffffu) << 16);
    g1[2] = (int)(((tensor_d0 >> 16) & 0xffffu) | ((tensor_d1 & 0xffffu) << 16));
    g1[3] = (int)(((tensor_d1 >> 16) & 0xffffu) | ((tile_d0 & 0xffffu) << 16));
    g1[4] = (int)(tile_d1 & 0xffffu);                          // tile_dim1 (dim2=0)
    g1[5] = (int)stride_elems;                                 // dim0_stride[31:0]
    g1[6] = 0;
    g1[7] = 0;
    v4i z4 = {0, 0, 0, 0};
#if __clang_major__ >= 23
    v8i z8 = {0, 0, 0, 0, 0, 0, 0, 0};
    __builtin_amdgcn_tensor_load_to_lds(g0, g1, z4, z4, z8, 0);
#else
    __builtin_amdgcn_tensor_load_to_lds(g0, g1, z4, z4, 0);
#endif
}
#endif

// ---------------------------------------------------------------------------
// f32 -> bf16 bulk weight conversion (once per launch)
// ---------------------------------------------------------------------------
__global__ void __launch_bounds__(256)
cvt_bf16(const float* __restrict__ src, ushort_t* __restrict__ dst)
{
    size_t i = ((size_t)blockIdx.x * 256 + threadIdx.x) * 4;
    float4 v = *(const float4*)(src + i);
    ushort_t pk[4] = { f2bf(v.x), f2bf(v.y), f2bf(v.z), f2bf(v.w) };
    *(ushort4*)(dst + i) = *(ushort4*)pk;
}

// ---------------------------------------------------------------------------
// GEMM: C[M,N] = act(A[M,K] * B[K,N] + bias[N])
//   A: f32 (Af32) or bf16 (Abf, double-buffered TDM), row-major lda=K
//   B: bf16 row-major ldb=N, staged transposed into LDS (double-buffered)
//   C: f32 (Cf) or bf16 (Cbf), ldc=N
// 256 threads = 8 waves; tile 128x128x32; wave = 32x64 = 2x4 WMMA frags.
// ---------------------------------------------------------------------------
__global__ void __launch_bounds__(256)
gemm_wmma_bf16(const float* __restrict__ Af32,
               const ushort_t* __restrict__ Abf,
               const ushort_t* __restrict__ Bb,
               const float* __restrict__ bias,
               float* __restrict__ Cf,
               ushort_t* __restrict__ Cbf,
               int M, int N, int K, int relu)
{
    __shared__ __align__(16) ushort_t a_lds[2][128 * APAD];
    __shared__ __align__(16) ushort_t b_lds[2][128 * APAD];  // transposed [col][k]

    const int tid  = threadIdx.x;
    const int wid  = tid >> 5;
    const int lane = tid & 31;
    const int half = lane >> 4;
    const int l16  = lane & 15;
    const int wm   = wid & 3;
    const int wn   = wid >> 2;
    const int m0   = blockIdx.y * 128;
    const int n0   = blockIdx.x * 128;

    v8f acc[2][4] = {};

    // ---- staging helpers ----
    auto stage_a = [&](int kk, ushort_t* abuf) {
        if (Abf) {
#if defined(HAVE_TDM)
            if (wid == 0)
                tdm_load_tile_bf16(Abf + (size_t)m0 * K + kk,
                                   (unsigned)(uintptr_t)abuf,
                                   (unsigned)K, (unsigned)M, BK, 128, (unsigned)K);
#else
            const int r = tid >> 2, c8 = tid & 3;
            #pragma unroll
            for (int p = 0; p < 2; ++p) {
                int row = r + 64 * p;
                const uint4 v = *(const uint4*)(Abf + (size_t)(m0 + row) * K + kk + c8 * 8);
                *(uint4*)(&abuf[row * APAD + c8 * 8]) = v;
            }
#endif
        } else {
            const int r = tid >> 3, c4 = tid & 7;
            #pragma unroll
            for (int p = 0; p < 4; ++p) {
                int row = r + 32 * p;
                const float4 v = *(const float4*)(Af32 + (size_t)(m0 + row) * K + kk + c4 * 4);
                ushort_t pk[4] = { f2bf(v.x), f2bf(v.y), f2bf(v.z), f2bf(v.w) };
                *(ushort4*)(&abuf[row * APAD + c4 * 4]) = *(ushort4*)pk;
            }
            if (kk + BK < K)
                __builtin_prefetch(Af32 + (size_t)(m0 + r) * K + kk + BK, 0, 0);
        }
    };
    auto stage_b = [&](int kk, ushort_t* bbuf) {
        const int k = tid >> 4, g = tid & 15;       // k row 0..15, col-group of 8
        #pragma unroll
        for (int p = 0; p < 2; ++p) {
            int kr = k + 16 * p;
            const uint4 v = *(const uint4*)(Bb + (size_t)(kk + kr) * N + n0 + g * 8);
            const unsigned w[4] = { v.x, v.y, v.z, v.w };
            #pragma unroll
            for (int j = 0; j < 4; ++j) {
                bbuf[(g * 8 + 2 * j + 0) * APAD + kr] = (ushort_t)(w[j] & 0xffffu);
                bbuf[(g * 8 + 2 * j + 1) * APAD + kr] = (ushort_t)(w[j] >> 16);
            }
        }
        if (kk + BK < K)
            __builtin_prefetch(Bb + (size_t)(kk + BK + k) * N + n0 + g * 8, 0, 0);
    };

    // ---- software pipeline: prime chunk 0 ----
    stage_a(0, a_lds[0]);
    stage_b(0, b_lds[0]);

    for (int kk = 0; kk < K; kk += BK) {
        const int ibuf  = (kk >> 5) & 1;
        const int knext = kk + BK;
        if (knext < K) {                     // issue next chunk early (async)
            stage_a(knext, a_lds[ibuf ^ 1]);
            stage_b(knext, b_lds[ibuf ^ 1]);
        }
#if defined(HAVE_TDM)
        if (Abf && wid == 0) {               // issue-early / wait-late
            if (knext < K) __builtin_amdgcn_s_wait_tensorcnt(1);
            else           __builtin_amdgcn_s_wait_tensorcnt(0);
        }
#endif
        __syncthreads();                     // publish chunk kk

        const ushort_t* al = a_lds[ibuf];
        const ushort_t* bl = b_lds[ibuf];
        FragBF fa[2];
        #pragma unroll
        for (int i = 0; i < 2; ++i) {
            const ushort_t* ar = &al[(32 * wm + 16 * i + l16) * APAD];
            fa[i].q[0] = *(const uint4*)(ar + 8 * half);        // K = 8h..8h+7
            fa[i].q[1] = *(const uint4*)(ar + 16 + 8 * half);   // K = 16+8h..
        }
        FragBF fb[4];
        #pragma unroll
        for (int j = 0; j < 4; ++j) {
            const ushort_t* br = &bl[(64 * wn + 16 * j + l16) * APAD];
            fb[j].q[0] = *(const uint4*)(br + 8 * half);        // K = 8h..8h+7
            fb[j].q[1] = *(const uint4*)(br + 16 + 8 * half);   // K = 16+8h..
        }
        #pragma unroll
        for (int i = 0; i < 2; ++i)
            #pragma unroll
            for (int j = 0; j < 4; ++j)
                acc[i][j] = __builtin_amdgcn_wmma_f32_16x16x32_bf16(
                    false, fa[i].v, false, fb[j].v, (short)0, acc[i][j], false, false);
        __syncthreads();                     // done reading before next overwrite
    }

    // ---- epilogue ----
    #pragma unroll
    for (int i = 0; i < 2; ++i) {
        #pragma unroll
        for (int j = 0; j < 4; ++j) {
            const int col = n0 + 64 * wn + 16 * j + l16;
            const float bv = bias[col];
            #pragma unroll
            for (int e = 0; e < 8; ++e) {
                const int row = m0 + 32 * wm + 16 * i + e + 8 * half;
                float c = acc[i][j][e] + bv;
                if (relu) c = fmaxf(c, 0.0f);
                if (Cbf) Cbf[(size_t)row * N + col] = f2bf(c);
                else     Cf [(size_t)row * N + col] = c;
            }
        }
    }
}

// ---------------------------------------------------------------------------
__global__ void __launch_bounds__(256)
init_h(const float* __restrict__ hidden, const unsigned char* __restrict__ dones,
       float* __restrict__ hm, ushort_t* __restrict__ h_bf)
{
    int idx = blockIdx.x * 256 + threadIdx.x;
    int b = idx >> 10;
    float v = dones[b] ? 0.0f : hidden[idx];
    hm[idx]   = v;
    h_bf[idx] = f2bf(v);
}

// ---------------------------------------------------------------------------
// One GRU step: 64 WGs x 256 threads; WG owns 16 columns of H; wave w owns
// batch rows 16w..16w+15; hh_r/z/n via WMMA over K=1024 with double-buffered
// TDM staging of the h_bf tile, fused gate math, in-place state update.
// ---------------------------------------------------------------------------
__global__ void __launch_bounds__(256)
gru_step(float* __restrict__ hm,
         ushort_t* __restrict__ h_bf,
         const float* __restrict__ xi,            // [T,B,3H]
         const unsigned char* __restrict__ dones, // [T,B]
         const ushort_t* __restrict__ Whb,        // [H,3H] bf16
         const float* __restrict__ bhn,
         ushort_t* __restrict__ y_bf,             // [T,B,H]
         float* __restrict__ h_out,               // [B,H]
         int t)
{
    __shared__ __align__(16) ushort_t a_lds[2][128 * APAD];
    __shared__ __align__(16) ushort_t b_lds[2][48 * APAD];   // [gate*16+cc][k]

    const int tid  = threadIdx.x;
    const int wid  = tid >> 5;
    const int lane = tid & 31;
    const int half = lane >> 4;
    const int l16  = lane & 15;
    const int n0   = blockIdx.x * 16;

    v8f accR = {}, accZ = {}, accN = {};

    auto stage_a = [&](int kk, ushort_t* abuf) {
#if defined(HAVE_TDM)
        if (wid == 0)
            tdm_load_tile_bf16(h_bf + kk, (unsigned)(uintptr_t)abuf,
                               HID, BATCH, BK, BATCH, HID);
#else
        const int r = tid >> 2, c8 = tid & 3;
        #pragma unroll
        for (int p = 0; p < 2; ++p) {
            int row = r + 64 * p;
            const uint4 v = *(const uint4*)(h_bf + (size_t)row * HID + kk + c8 * 8);
            *(uint4*)(&abuf[row * APAD + c8 * 8]) = v;
        }
#endif
    };
    auto stage_b = [&](int kk, ushort_t* bbuf) {
        #pragma unroll
        for (int e = 0; e < 6; ++e) {
            int idx = tid * 6 + e;           // 0..1535
            int k = idx & 31, c = idx >> 5;  // c = gate*16 + cc
            bbuf[c * APAD + k] =
                Whb[(size_t)(kk + k) * 3072 + (c >> 4) * HID + n0 + (c & 15)];
        }
        if (kk + BK < HID)
            __builtin_prefetch(Whb + (size_t)(kk + BK) * 3072 + n0, 0, 1);
    };

    stage_a(0, a_lds[0]);
    stage_b(0, b_lds[0]);

    for (int kk = 0; kk < HID; kk += BK) {
        const int ibuf  = (kk >> 5) & 1;
        const int knext = kk + BK;
        if (knext < HID) {
            stage_a(knext, a_lds[ibuf ^ 1]);
            stage_b(knext, b_lds[ibuf ^ 1]);
        }
#if defined(HAVE_TDM)
        if (wid == 0) {
            if (knext < HID) __builtin_amdgcn_s_wait_tensorcnt(1);
            else             __builtin_amdgcn_s_wait_tensorcnt(0);
        }
#endif
        __syncthreads();

        const ushort_t* al = a_lds[ibuf];
        const ushort_t* bl = b_lds[ibuf];
        FragBF fa;
        const ushort_t* ar = &al[(16 * wid + l16) * APAD];
        fa.q[0] = *(const uint4*)(ar + 8 * half);
        fa.q[1] = *(const uint4*)(ar + 16 + 8 * half);

        FragBF fr, fz, fn;
        #pragma unroll
        for (int g = 0; g < 3; ++g) {
            const ushort_t* br = &bl[(g * 16 + l16) * APAD];
            FragBF f;
            f.q[0] = *(const uint4*)(br + 8 * half);
            f.q[1] = *(const uint4*)(br + 16 + 8 * half);
            if (g == 0) fr = f; else if (g == 1) fz = f; else fn = f;
        }
        accR = __builtin_amdgcn_wmma_f32_16x16x32_bf16(false, fa.v, false, fr.v, (short)0, accR, false, false);
        accZ = __builtin_amdgcn_wmma_f32_16x16x32_bf16(false, fa.v, false, fz.v, (short)0, accZ, false, false);
        accN = __builtin_amdgcn_wmma_f32_16x16x32_bf16(false, fa.v, false, fn.v, (short)0, accN, false, false);
        __syncthreads();
    }

    // ---- fused GRU gate math ----
    const int c  = n0 + l16;
    const float bn = bhn[c];
    #pragma unroll
    for (int e = 0; e < 8; ++e) {
        const int b = 16 * wid + e + 8 * half;
        const size_t xb = ((size_t)t * BATCH + b) * 3072;
        const float xr = xi[xb + c];
        const float xz = xi[xb + HID + c];
        const float xn = xi[xb + 2 * HID + c];
        const float hold = hm[(size_t)b * HID + c];
        const float r = sigmoidf_(xr + accR[e]);
        const float z = sigmoidf_(xz + accZ[e]);
        const float n = tanhf(xn + r * (accN[e] + bn));
        const float hnew = (1.0f - z) * n + z * hold;

        y_bf[((size_t)t * BATCH + b) * HID + c] = f2bf(hnew);
        if (t == T_STEPS - 1) h_out[(size_t)b * HID + c] = hnew;

        const float hmask = (t + 1 < T_STEPS && dones[(t + 1) * BATCH + b]) ? 0.0f : hnew;
        hm  [(size_t)b * HID + c] = hmask;
        h_bf[(size_t)b * HID + c] = f2bf(hmask);
    }
}

// ---------------------------------------------------------------------------
__global__ void __launch_bounds__(256)
value_head(const ushort_t* __restrict__ critic, const float* __restrict__ W2,
           const float* __restrict__ b2, float* __restrict__ value)
{
    const int row  = blockIdx.x * 8 + (threadIdx.x >> 5);
    const int lane = threadIdx.x & 31;
    float sum = 0.0f;
    for (int c = lane; c < HID; c += 32)
        sum += bf2f(critic[(size_t)row * HID + c]) * W2[c];
    #pragma unroll
    for (int off = 16; off > 0; off >>= 1)
        sum += __shfl_down(sum, off, 32);
    if (lane == 0) value[row] = sum + b2[0];
}

// ---------------------------------------------------------------------------
extern "C" void kernel_launch(void* const* d_in, const int* in_sizes, int n_in,
                              void* d_out, int out_size, void* d_ws, size_t ws_size,
                              hipStream_t stream)
{
    const float*         hidden = (const float*)d_in[0];
    const float*         ws     = (const float*)d_in[1];
    const unsigned char* dones  = (const unsigned char*)d_in[2];
    const float*         W_emb  = (const float*)d_in[3];
    const float*         b_emb  = (const float*)d_in[4];
    const float*         Wi     = (const float*)d_in[5];
    const float*         bi     = (const float*)d_in[6];
    const float*         Wh     = (const float*)d_in[7];
    const float*         bhn    = (const float*)d_in[8];
    const float*         W1     = (const float*)d_in[9];
    const float*         b1     = (const float*)d_in[10];
    const float*         W2     = (const float*)d_in[11];
    const float*         b2     = (const float*)d_in[12];

    float* h_out = (float*)d_out;                        // [B,H]
    float* value = (float*)d_out + (size_t)BATCH * HID;  // [T,B]

    // ---- workspace layout ----
    char* p = (char*)d_ws;
    const size_t M = (size_t)T_STEPS * BATCH;            // 16384
    float*    xi      = (float*)p;     p += M * 3 * HID * sizeof(float);
    ushort_t* emb_bf  = (ushort_t*)p;                    // aliased with critic
    ushort_t* critic  = emb_bf;
    p += M * HID * sizeof(ushort_t);
    ushort_t* y_bf    = (ushort_t*)p;  p += M * HID * sizeof(ushort_t);
    float*    hm      = (float*)p;     p += (size_t)BATCH * HID * sizeof(float);
    ushort_t* h_bf    = (ushort_t*)p;  p += (size_t)BATCH * HID * sizeof(ushort_t);
    ushort_t* W_emb_b = (ushort_t*)p;  p += (size_t)OBS_DIM * HID * sizeof(ushort_t);
    ushort_t* Wi_b    = (ushort_t*)p;  p += (size_t)HID * 3 * HID * sizeof(ushort_t);
    ushort_t* Wh_b    = (ushort_t*)p;  p += (size_t)HID * 3 * HID * sizeof(ushort_t);
    ushort_t* W1_b    = (ushort_t*)p;

    // 0) one-time weight conversion to bf16
    cvt_bf16<<<(OBS_DIM * HID) / 1024, 256, 0, stream>>>(W_emb, W_emb_b);
    cvt_bf16<<<(HID * 3 * HID) / 1024, 256, 0, stream>>>(Wi, Wi_b);
    cvt_bf16<<<(HID * 3 * HID) / 1024, 256, 0, stream>>>(Wh, Wh_b);
    cvt_bf16<<<(HID * HID) / 1024, 256, 0, stream>>>(W1, W1_b);

    // 1) emb = relu(world_state @ W_emb + b_emb) -> bf16 [16384,1024]
    gemm_wmma_bf16<<<dim3(HID / 128, M / 128), 256, 0, stream>>>(
        ws, nullptr, W_emb_b, b_emb, nullptr, emb_bf, (int)M, HID, OBS_DIM, 1);

    // 2) xi = emb @ Wi + bi -> f32 [16384,3072]
    gemm_wmma_bf16<<<dim3(3 * HID / 128, M / 128), 256, 0, stream>>>(
        nullptr, emb_bf, Wi_b, bi, xi, nullptr, (int)M, 3 * HID, HID, 0);

    // 3) init masked hidden state
    init_h<<<(BATCH * HID) / 256, 256, 0, stream>>>(hidden, dones, hm, h_bf);

    // 4) recurrent scan (graph-captured per-step launches)
    for (int t = 0; t < T_STEPS; ++t)
        gru_step<<<HID / 16, 256, 0, stream>>>(hm, h_bf, xi, dones, Wh_b, bhn,
                                               y_bf, h_out, t);

    // 5) critic = relu(y @ W1 + b1) -> bf16 [16384,1024]
    gemm_wmma_bf16<<<dim3(HID / 128, M / 128), 256, 0, stream>>>(
        nullptr, y_bf, W1_b, b1, nullptr, critic, (int)M, HID, HID, 1);

    // 6) value head
    value_head<<<M / 8, 256, 0, stream>>>(critic, W2, b2, value);
}
